// SelfAttention_65481071406758
// MI455X (gfx1250) — matrix operand order repaired and checked by reading, not compiled, not run on hardware
//
#include <hip/hip_runtime.h>

#define EMBED   1024
#define NHEADS  16
#define HDIM    64
#define BATCH   2
#define SEQ     2048
#define ROWS    (BATCH * SEQ)      // 4096
#define QKV_COLS (3 * EMBED)       // 3072
#define ATTN_SCALE 0.125f          // 1/sqrt(64)

typedef __bf16 bf16x16 __attribute__((ext_vector_type(16)));
typedef __bf16 bf16x8  __attribute__((ext_vector_type(8)));
typedef float  f32x8   __attribute__((ext_vector_type(8)));

static __device__ __forceinline__ unsigned short f32_to_bf16_raw(float f) {
    union { float f; unsigned int u; } v; v.f = f;
    unsigned int r = 0x7FFFu + ((v.u >> 16) & 1u);   // round-to-nearest-even
    return (unsigned short)((v.u + r) >> 16);
}

static __device__ __forceinline__ f32x8 zero8() {
    f32x8 z = {0.f,0.f,0.f,0.f,0.f,0.f,0.f,0.f};
    return z;
}

static __device__ __forceinline__ f32x8 wmma_bf16(bf16x16 a, bf16x16 b, f32x8 c) {
    return __builtin_amdgcn_wmma_f32_16x16x32_bf16(
        /*neg_a=*/false, a, /*neg_b=*/false, b,
        /*c_mod=*/(short)0, c, /*reuse_a=*/false, /*reuse_b=*/false);
}

// A-fragment (16x32 bf16, MxK): lane holds row = lane%16; lane half selects
// contiguous K run [half*16, half*16+16) (matches B fragment order)
static __device__ __forceinline__ bf16x16 load_a_frag(const unsigned short* rowPtr, int half) {
    bf16x8 lo = *(const bf16x8*)(rowPtr + half * 16);
    bf16x8 hi = *(const bf16x8*)(rowPtr + half * 16 + 8);
    return __builtin_shufflevector(lo, hi, 0,1,2,3,4,5,6,7,8,9,10,11,12,13,14,15);
}

// B-fragment (32x16 bf16, KxN): lane holds col = lane%16; contiguous K run
// [half*16, half*16+16). colPtr points at the start of that column's K row.
static __device__ __forceinline__ bf16x16 load_b_frag(const unsigned short* colPtr, int half) {
    return *(const bf16x16*)(colPtr + half * 16);
}

// ---------------------------------------------------------------------------
// Kernel 1: fp32 -> bf16, 4-wide vectorized (pure bandwidth)
// ---------------------------------------------------------------------------
__global__ __launch_bounds__(256) void cvt_f32_bf16_x4(
    const float* __restrict__ in, unsigned short* __restrict__ out, int n4) {
    int i = blockIdx.x * blockDim.x + threadIdx.x;
    int stride = gridDim.x * blockDim.x;
    for (; i < n4; i += stride) {
        float4 v = ((const float4*)in)[i];
        ushort4 o;
        o.x = f32_to_bf16_raw(v.x);
        o.y = f32_to_bf16_raw(v.y);
        o.z = f32_to_bf16_raw(v.z);
        o.w = f32_to_bf16_raw(v.w);
        ((ushort4*)out)[i] = o;
    }
}

// ---------------------------------------------------------------------------
// GEMM core: 16 rows x 64 cols per wave, K = EMBED, bf16 WMMA with explicit
// ping-pong double buffering. The k-loop is explicitly NOT unrolled further:
// compiler unrolling on top of the manual pipeline blows up VGPR tuple
// pressure and makes the allocator split accumulator live ranges
// (v_mov_b64 + hazard v_nop chains after every WMMA).
// ---------------------------------------------------------------------------
struct FragSet {
    bf16x16 a, b0, b1, b2, b3;
};

static __device__ __forceinline__ void load_set(
    FragSet& s, int kk, int half,
    const unsigned short* aRow, const unsigned short* c0,
    const unsigned short* c1, const unsigned short* c2, const unsigned short* c3) {
    s.a  = load_a_frag(aRow + kk, half);
    s.b0 = load_b_frag(c0 + kk, half);
    s.b1 = load_b_frag(c1 + kk, half);
    s.b2 = load_b_frag(c2 + kk, half);
    s.b3 = load_b_frag(c3 + kk, half);
}

static __device__ __forceinline__ void mma_set(f32x8 (&acc)[4], const FragSet& s) {
    acc[0] = wmma_bf16(s.a, s.b0, acc[0]);
    acc[1] = wmma_bf16(s.a, s.b1, acc[1]);
    acc[2] = wmma_bf16(s.a, s.b2, acc[2]);
    acc[3] = wmma_bf16(s.a, s.b3, acc[3]);
}

static __device__ __forceinline__ void gemm_16x64(
    f32x8 (&acc)[4], int half,
    const unsigned short* aRow, const unsigned short* c0,
    const unsigned short* c1, const unsigned short* c2, const unsigned short* c3) {
    FragSet s0, s1;                       // ping-pong buffers
    load_set(s0, 0,  half, aRow, c0, c1, c2, c3);
    load_set(s1, 32, half, aRow, c0, c1, c2, c3);
    // chunks processed: kk and kk+32 per body; bodies at kk = 0,64,...,EMBED-128
    #pragma clang loop unroll(disable)
    for (int kk = 0; kk < EMBED - 64; kk += 64) {
        mma_set(acc, s0);
        load_set(s0, kk + 64, half, aRow, c0, c1, c2, c3);
        mma_set(acc, s1);
        load_set(s1, kk + 96, half, aRow, c0, c1, c2, c3);
    }
    mma_set(acc, s0);                     // chunk EMBED-64
    mma_set(acc, s1);                     // chunk EMBED-32
}

// ---------------------------------------------------------------------------
// Kernel 2: QKV projection  C[4096,3072] = Xbf[4096,1024] * Wqkv^T
// Results scattered into q[b,h,n,d], k[b,h,n,d], vt[b,h,d,n] (bf16).
// ---------------------------------------------------------------------------
__global__ __launch_bounds__(256) void qkv_gemm(
    const unsigned short* __restrict__ xb,
    const unsigned short* __restrict__ wb,
    unsigned short* __restrict__ q,
    unsigned short* __restrict__ k,
    unsigned short* __restrict__ vt) {
    const int lane = threadIdx.x & 31;
    const int wave = threadIdx.x >> 5;
    const int half = lane >> 4;
    const int l16  = lane & 15;

    const int row0 = blockIdx.x * 128 + wave * 16;   // 32 blocks * 128 = 4096
    const int col0 = blockIdx.y * 64;                // 48 blocks * 64  = 3072

    f32x8 acc[4] = { zero8(), zero8(), zero8(), zero8() };

    const unsigned short* aRow = xb + (size_t)(row0 + l16) * EMBED;
    const unsigned short* c0   = wb + (size_t)(col0 +  0 + l16) * EMBED;
    const unsigned short* c1   = wb + (size_t)(col0 + 16 + l16) * EMBED;
    const unsigned short* c2   = wb + (size_t)(col0 + 32 + l16) * EMBED;
    const unsigned short* c3   = wb + (size_t)(col0 + 48 + l16) * EMBED;

    gemm_16x64(acc, half, aRow, c0, c1, c2, c3);

    #pragma unroll
    for (int t = 0; t < 4; t++) {
        int col   = col0 + t * 16 + l16;
        int which = col >> 10;       // 0=q 1=k 2=v
        int rcol  = col & 1023;
        int h     = rcol >> 6;
        int d     = rcol & 63;
        #pragma unroll
        for (int rr = 0; rr < 8; rr++) {
            int row = row0 + rr + half * 8;
            int bb  = row >> 11;
            int n   = row & (SEQ - 1);
            unsigned short val = f32_to_bf16_raw(acc[t][rr]);
            size_t bh = (size_t)(bb * NHEADS + h);
            if (which == 0)      q [(bh * SEQ + n) * HDIM + d] = val;
            else if (which == 1) k [(bh * SEQ + n) * HDIM + d] = val;
            else                 vt[(bh * HDIM + d) * SEQ + n] = val;
        }
    }
}

// ---------------------------------------------------------------------------
// Kernel 3: flash attention. 4 waves/block, each wave owns a 16-query tile.
// Online softmax across 2048 keys in chunks of 32; P converted C->A layout
// through a padded LDS tile (stride 40 u16 = 80B, conflict-free, 16B aligned).
// V-fragments are issued before the softmax VALU section so their latency is
// hidden; next chunk's K rows get a global_prefetch.
// ---------------------------------------------------------------------------
__global__ __launch_bounds__(128) void flash_attn(
    const unsigned short* __restrict__ q,
    const unsigned short* __restrict__ k,
    const unsigned short* __restrict__ vt,
    unsigned short* __restrict__ o) {          // [4096, 1024] bf16
    __shared__ __align__(16) unsigned short pTile[4][16 * 40];

    const int lane = threadIdx.x & 31;
    const int wave = threadIdx.x >> 5;
    const int half = lane >> 4;
    const int l16  = lane & 15;

    const int bh    = blockIdx.x >> 5;            // b*16 + h
    const int qc    = blockIdx.x & 31;
    const int qbase = qc * 64 + wave * 16;

    const unsigned short* qBase  = q  + (size_t)bh * SEQ * HDIM;
    const unsigned short* kBase  = k  + (size_t)bh * SEQ * HDIM;
    const unsigned short* vtBase = vt + (size_t)bh * HDIM * SEQ;

    const unsigned short* qRow = qBase + (size_t)(qbase + l16) * HDIM;
    bf16x16 qa0 = load_a_frag(qRow,      half);   // d 0..31
    bf16x16 qa1 = load_a_frag(qRow + 32, half);   // d 32..63

    f32x8 oAcc[4] = { zero8(), zero8(), zero8(), zero8() };
    float rowMax[8], rowSum[8];
    #pragma unroll
    for (int r = 0; r < 8; r++) { rowMax[r] = -1e30f; rowSum[r] = 0.f; }

    unsigned short* pw = pTile[wave];

    #pragma clang loop unroll(disable)
    for (int m0 = 0; m0 < SEQ; m0 += 32) {
        // ---- scores: S = Q * K^T for 32 keys (two 16-col tiles) ----
        const unsigned short* kRow0 = kBase + (size_t)(m0      + l16) * HDIM;
        const unsigned short* kRow1 = kBase + (size_t)(m0 + 16 + l16) * HDIM;
        f32x8 s0 = zero8(), s1 = zero8();
        s0 = wmma_bf16(qa0, load_b_frag(kRow0,      half), s0);
        s0 = wmma_bf16(qa1, load_b_frag(kRow0 + 32, half), s0);
        s1 = wmma_bf16(qa0, load_b_frag(kRow1,      half), s1);
        s1 = wmma_bf16(qa1, load_b_frag(kRow1 + 32, half), s1);

        // ---- issue V fragments now; softmax VALU below hides their latency
        bf16x16 vb0 = *(const bf16x16*)(vtBase + (size_t)( 0 + l16) * SEQ + m0 + half * 16);
        bf16x16 vb1 = *(const bf16x16*)(vtBase + (size_t)(16 + l16) * SEQ + m0 + half * 16);
        bf16x16 vb2 = *(const bf16x16*)(vtBase + (size_t)(32 + l16) * SEQ + m0 + half * 16);
        bf16x16 vb3 = *(const bf16x16*)(vtBase + (size_t)(48 + l16) * SEQ + m0 + half * 16);

        // warm next chunk's K rows in cache (global_prefetch, no VGPR return)
        if (m0 + 32 < SEQ) {
            __builtin_prefetch(kBase + (size_t)(m0 + 32 + l16) * HDIM, 0, 3);
            __builtin_prefetch(kBase + (size_t)(m0 + 48 + l16) * HDIM, 0, 3);
        }

        // ---- online softmax (per-row reductions inside 16-lane halves) ----
        #pragma unroll
        for (int r = 0; r < 8; r++) {
            float sa = s0[r] * ATTN_SCALE;
            float sb = s1[r] * ATTN_SCALE;
            float mx = fmaxf(sa, sb);
            mx = fmaxf(mx, __shfl_xor(mx, 1, 32));
            mx = fmaxf(mx, __shfl_xor(mx, 2, 32));
            mx = fmaxf(mx, __shfl_xor(mx, 4, 32));
            mx = fmaxf(mx, __shfl_xor(mx, 8, 32));
            float newMax = fmaxf(rowMax[r], mx);
            float corr   = __expf(rowMax[r] - newMax);
            float pa     = __expf(sa - newMax);
            float pb     = __expf(sb - newMax);
            float ps     = pa + pb;
            ps += __shfl_xor(ps, 1, 32);
            ps += __shfl_xor(ps, 2, 32);
            ps += __shfl_xor(ps, 4, 32);
            ps += __shfl_xor(ps, 8, 32);
            rowSum[r] = rowSum[r] * corr + ps;
            rowMax[r] = newMax;
            #pragma unroll
            for (int dt = 0; dt < 4; dt++) oAcc[dt][r] *= corr;
            int row = r + half * 8;
            pw[row * 40      + l16] = f32_to_bf16_raw(pa);
            pw[row * 40 + 16 + l16] = f32_to_bf16_raw(pb);
        }

        // ---- re-read P as an A-fragment (16x32 over keys) ----
        // (same-wave LDS ops are in-order; DScnt waits inserted by compiler)
        const unsigned short* pr = pw + l16 * 40;
        bf16x8 plo = *(const bf16x8*)(pr + half * 16);
        bf16x8 phi = *(const bf16x8*)(pr + half * 16 + 8);
        bf16x16 pfrag = __builtin_shufflevector(plo, phi,
                            0,1,2,3,4,5,6,7,8,9,10,11,12,13,14,15);

        // ---- O += P * V  (V pre-transposed -> contiguous B fragments) ----
        oAcc[0] = wmma_bf16(pfrag, vb0, oAcc[0]);
        oAcc[1] = wmma_bf16(pfrag, vb1, oAcc[1]);
        oAcc[2] = wmma_bf16(pfrag, vb2, oAcc[2]);
        oAcc[3] = wmma_bf16(pfrag, vb3, oAcc[3]);
    }

    // ---- normalize + store attn output (bf16, [b*N+n, h*64+d]) ----
    const int b    = bh >> 4;
    const int hOff = (bh & 15) * HDIM;
    #pragma unroll
    for (int dt = 0; dt < 4; dt++) {
        #pragma unroll
        for (int r = 0; r < 8; r++) {
            int n = qbase + r + half * 8;
            float val = oAcc[dt][r] / rowSum[r];
            o[(size_t)(b * SEQ + n) * EMBED + hOff + dt * 16 + l16] = f32_to_bf16_raw(val);
        }
    }
}

// ---------------------------------------------------------------------------
// Kernel 4: output projection  out[4096,1024] = Attn(bf16) * Wout^T + bias
// ---------------------------------------------------------------------------
__global__ __launch_bounds__(256) void out_gemm(
    const unsigned short* __restrict__ ab,
    const unsigned short* __restrict__ wb,
    const float* __restrict__ bias,
    float* __restrict__ out) {
    const int lane = threadIdx.x & 31;
    const int wave = threadIdx.x >> 5;
    const int half = lane >> 4;
    const int l16  = lane & 15;

    const int row0 = blockIdx.x * 128 + wave * 16;   // 32 blocks
    const int col0 = blockIdx.y * 64;                // 16 blocks

    f32x8 acc[4] = { zero8(), zero8(), zero8(), zero8() };

    const unsigned short* aRow = ab + (size_t)(row0 + l16) * EMBED;
    const unsigned short* c0   = wb + (size_t)(col0 +  0 + l16) * EMBED;
    const unsigned short* c1   = wb + (size_t)(col0 + 16 + l16) * EMBED;
    const unsigned short* c2   = wb + (size_t)(col0 + 32 + l16) * EMBED;
    const unsigned short* c3   = wb + (size_t)(col0 + 48 + l16) * EMBED;

    gemm_16x64(acc, half, aRow, c0, c1, c2, c3);

    #pragma unroll
    for (int t = 0; t < 4; t++) {
        int col = col0 + t * 16 + l16;
        float bv = bias[col];
        #pragma unroll
        for (int rr = 0; rr < 8; rr++) {
            int row = row0 + rr + half * 8;
            out[(size_t)row * EMBED + col] = acc[t][rr] + bv;
        }
    }
}

// ---------------------------------------------------------------------------
extern "C" void kernel_launch(void* const* d_in, const int* in_sizes, int n_in,
                              void* d_out, int out_size, void* d_ws, size_t ws_size,
                              hipStream_t stream) {
    const float* x     = (const float*)d_in[0];
    const float* w_qkv = (const float*)d_in[1];
    const float* w_out = (const float*)d_in[2];
    const float* b_out = (const float*)d_in[3];

    char* ws = (char*)d_ws;
    size_t off = 0;
    auto carve = [&](size_t bytes) -> void* {
        void* p = ws + off;
        off += (bytes + 255) & ~(size_t)255;
        return p;
    };
    unsigned short* xb      = (unsigned short*)carve((size_t)ROWS * EMBED * 2);
    unsigned short* wqkvb   = (unsigned short*)carve((size_t)QKV_COLS * EMBED * 2);
    unsigned short* woutb   = (unsigned short*)carve((size_t)EMBED * EMBED * 2);
    unsigned short* qbuf    = (unsigned short*)carve((size_t)BATCH * NHEADS * SEQ * HDIM * 2);
    unsigned short* kbuf    = (unsigned short*)carve((size_t)BATCH * NHEADS * SEQ * HDIM * 2);
    unsigned short* vtbuf   = (unsigned short*)carve((size_t)BATCH * NHEADS * HDIM * SEQ * 2);
    unsigned short* attnOut = (unsigned short*)carve((size_t)ROWS * EMBED * 2);
    if (off > ws_size) return;  // workspace too small: bail deterministically

    auto cvt = [&](const float* src, unsigned short* dst, int n) {
        int n4 = n / 4;
        int blocks = (n4 + 255) / 256;
        if (blocks > 2048) blocks = 2048;
        cvt_f32_bf16_x4<<<blocks, 256, 0, stream>>>(src, dst, n4);
    };
    cvt(x,     xb,    ROWS * EMBED);
    cvt(w_qkv, wqkvb, EMBED * QKV_COLS);
    cvt(w_out, woutb, EMBED * EMBED);

    qkv_gemm<<<dim3(ROWS / 128, QKV_COLS / 64), 256, 0, stream>>>(
        xb, wqkvb, qbuf, kbuf, vtbuf);

    flash_attn<<<BATCH * NHEADS * (SEQ / 64), 128, 0, stream>>>(
        qbuf, kbuf, vtbuf, attnOut);

    out_gemm<<<dim3(ROWS / 128, EMBED / 64), 256, 0, stream>>>(
        attnOut, woutb, b_out, (float*)d_out);
}
